// STPGCNs_2130303779460
// MI455X (gfx1250) — matrix operand order repaired and checked by reading, not compiled, not run on hardware
//
#include <hip/hip_runtime.h>
#include <hip/hip_bf16.h>

// Problem constants (match reference)
#define B_   4
#define T_   12
#define V_   512
#define C_   64
#define D_   8
#define TS_  3
#define L_   3
#define P_   12
#define C2_  128          // 2C
#define C4_  256          // 4C
#define BTVC (B_*T_*V_*C_)

typedef __attribute__((ext_vector_type(16))) __bf16 v16bf;
typedef __attribute__((ext_vector_type(8)))  float  v8f;
typedef __attribute__((ext_vector_type(4)))  unsigned int v4u;
typedef __attribute__((ext_vector_type(8)))  int  v8i_;
typedef __attribute__((ext_vector_type(4)))  int  v4i_;

union BF16x16 { unsigned int u[8]; v16bf v; };

__device__ __forceinline__ unsigned short f2bf(float f) {
    unsigned int u = __float_as_uint(f);
    u += 0x7FFFu + ((u >> 16) & 1u);       // round-to-nearest-even
    return (unsigned short)(u >> 16);
}
__device__ __forceinline__ float sigmoidf_(float x) {
    return 1.0f / (1.0f + __expf(-x));
}

// ---------------------------------------------------------------------------
// h0 = x @ W_in + b_in  ->  bf16 in both [b][t][v][c] and [b][t][c][v] layouts
__global__ void k_prep_h(const float* __restrict__ x, const float* __restrict__ W_in,
                         const float* __restrict__ b_in,
                         unsigned short* __restrict__ h, unsigned short* __restrict__ hT) {
    int idx = blockIdx.x * blockDim.x + threadIdx.x;
    if (idx >= BTVC) return;
    int c  = idx % C_;
    int v  = (idx / C_) % V_;
    int bt = idx / (C_ * V_);
    float val = x[bt * V_ + v] * W_in[c] + b_in[c];
    unsigned short bv = f2bf(val);
    h[idx] = bv;
    hT[((size_t)bt * C_ + c) * V_ + v] = bv;
}

// generic f32 -> bf16 conversion
__global__ void k_f2bf_arr(const float* __restrict__ src, unsigned short* __restrict__ dst, int n) {
    int i = blockIdx.x * blockDim.x + threadIdx.x;
    if (i < n) dst[i] = f2bf(src[i]);
}

// ---------------------------------------------------------------------------
// Emask[tau][vj][vi] = exp(spat[vi,vj]) * range_mask[vj, tau*V+vi]   (bf16)
__global__ void k_emask(const float* __restrict__ sape, const float* __restrict__ srpe,
                        const float* __restrict__ mask, const float* __restrict__ ri_mu,
                        const float* __restrict__ ri_sigma, int layer,
                        unsigned short* __restrict__ emask) {
    int idx = blockIdx.x * blockDim.x + threadIdx.x;   // V*V
    if (idx >= V_ * V_) return;
    int vi = idx % V_;
    int vj = idx / V_;
    const float* mu = ri_mu    + layer * 6 * D_;
    const float* sg = ri_sigma + layer * 6 * D_;
    float s = 0.f;
    #pragma unroll
    for (int dd = 0; dd < D_; ++dd) {
        float e0 = sape[vi * D_ + dd] - mu[0 * D_ + dd]; float g0 = sg[0 * D_ + dd];
        float e1 = sape[vj * D_ + dd] - mu[1 * D_ + dd]; float g1 = sg[1 * D_ + dd];
        float e4 = srpe[((size_t)vi * V_ + vj) * D_ + dd] - mu[4 * D_ + dd];
        float g4 = sg[4 * D_ + dd];
        s += -0.5f * (e0 * e0 * g0 * g0 + e1 * e1 * g1 * g1 + e4 * e4 * g4 * g4);
    }
    float e = __expf(s);
    for (int tau = 0; tau < TS_; ++tau) {
        float mv = mask[(size_t)vj * (TS_ * V_) + tau * V_ + vi];
        emask[((size_t)tau * V_ + vj) * V_ + vi] = f2bf(e * mv);
    }
}

// Wfac[b][t][tau] = exp(g_ti + g_tj + g_tr)
__global__ void k_wfac(const float* __restrict__ tape, const float* __restrict__ trpe,
                       const float* __restrict__ ri_mu, const float* __restrict__ ri_sigma,
                       int layer, float* __restrict__ wfac) {
    int idx = threadIdx.x;
    if (idx >= B_ * T_ * TS_) return;
    int tau = idx % TS_;
    int t   = (idx / TS_) % T_;
    int b   = idx / (TS_ * T_);
    const float* mu = ri_mu    + layer * 6 * D_;
    const float* sg = ri_sigma + layer * 6 * D_;
    int src = t + tau - (TS_ - 1);
    float s = 0.f;
    #pragma unroll
    for (int dd = 0; dd < D_; ++dd) {
        float ti = tape[(b * T_ + t) * D_ + dd] - mu[2 * D_ + dd];
        float g2 = sg[2 * D_ + dd];
        float tjv = (src >= 0) ? tape[(b * T_ + src) * D_ + dd] : 0.f;
        float tj = tjv - mu[3 * D_ + dd];
        float g3 = sg[3 * D_ + dd];
        float tr = trpe[tau * D_ + dd] - mu[5 * D_ + dd];
        float g5 = sg[5 * D_ + dd];
        s += -0.5f * (ti * ti * g2 * g2 + tj * tj * g3 * g3 + tr * tr * g5 * g5);
    }
    wfac[idx] = __expf(s);
}

// SE[vj][o] = sum_d sape[vj][d] * gc_sew[layer][o][d]
__global__ void k_se(const float* __restrict__ sape, const float* __restrict__ gc_sew,
                     int layer, float* __restrict__ se) {
    int idx = blockIdx.x * blockDim.x + threadIdx.x;   // V*2C
    if (idx >= V_ * C2_) return;
    int o = idx % C2_, vj = idx / C2_;
    float acc = 0.f;
    #pragma unroll
    for (int dd = 0; dd < D_; ++dd)
        acc += sape[vj * D_ + dd] * gc_sew[(layer * C2_ + o) * D_ + dd];
    se[idx] = acc;
}

// TE[b][t][o] = sum_d tape[b][t][0][d] * gc_tew[layer][o][d]
__global__ void k_te(const float* __restrict__ tape, const float* __restrict__ gc_tew,
                     int layer, float* __restrict__ te) {
    int idx = blockIdx.x * blockDim.x + threadIdx.x;   // B*T*2C
    if (idx >= B_ * T_ * C2_) return;
    int o = idx % C2_, bt = idx / C2_;
    float acc = 0.f;
    #pragma unroll
    for (int dd = 0; dd < D_; ++dd)
        acc += tape[bt * D_ + dd] * gc_tew[(layer * C2_ + o) * D_ + dd];
    te[idx] = acc;
}

// ---------------------------------------------------------------------------
// Fused layer: agg = sum_tau Wfac * (Emask_tau @ H_{t+tau}); z = agg@xw^T + biases;
// layernorm; GLU -> new h (bf16, both layouts).
// A operand (Emask tile: 64 contiguous rows x 512 K = 64 KB) is DMA'd to LDS by
// the Tensor Data Mover (tensor_load_to_lds + s_wait_tensorcnt), then read as
// WMMA A-fragments via LDS. B (h^T) streams from global (L2 resident).
__global__ __launch_bounds__(128) void k_layer(
        const unsigned short* __restrict__ emask, const unsigned short* __restrict__ hT_in,
        const float* __restrict__ wfac, const float* __restrict__ se,
        const float* __restrict__ te, const float* __restrict__ gc_xw,
        const float* __restrict__ gc_xb, const float* __restrict__ gc_ln_g,
        const float* __restrict__ gc_ln_b, int layer,
        unsigned short* __restrict__ h_out, unsigned short* __restrict__ hT_out) {
    __shared__ unsigned short aL[64][V_];     // 64 KB Emask tile (TDM destination)
    __shared__ float aggL[64][C_ + 4];
    __shared__ float xwL[C2_][C_];
    __shared__ float zL[64][C2_];
    __shared__ float redS[64][2];
    __shared__ float redQ[64][2];
    const v8f vzero = {0, 0, 0, 0, 0, 0, 0, 0};

    int b = blockIdx.z, t = blockIdx.y;
    int vjblk = blockIdx.x * 64;
    int tid = threadIdx.x;
    int lane = tid & 31, wave = tid >> 5;
    int m = lane & 15, hlf = lane >> 4;

    for (int i = tid; i < C2_ * C_; i += 128)
        xwL[i / C_][i % C_] = gc_xw[layer * C2_ * C_ + i];

    v8f facc[4];
    #pragma unroll
    for (int n = 0; n < 4; ++n) facc[n] = vzero;

    unsigned int ldsoff = (unsigned int)(unsigned long long)(&aL[0][0]);

    for (int tau = 0; tau < TS_; ++tau) {
        int ts = t + tau - (TS_ - 1);
        if (ts < 0) continue;                 // zero-padded h rows: exact skip
        float wf = wfac[(b * T_ + t) * TS_ + tau];

        __syncthreads();                      // previous tile fully consumed
        if (wave == 0) {
            // Tensor DMA descriptor (D#), ISA 8.3/8.4: 2D tile 512x64, 2-byte elems
            unsigned long long ga =
                (unsigned long long)(const void*)(emask + ((size_t)tau * V_ + vjblk) * V_);
            v4u g0;
            g0[0] = 1u;                                       // count=1, user mode
            g0[1] = ldsoff;                                   // lds_addr
            g0[2] = (unsigned int)(ga & 0xFFFFFFFFull);       // global_addr[31:0]
            g0[3] = (unsigned int)((ga >> 32) & 0x01FFFFFFull)
                  | (2u << 30);                               // global_addr[56:32] | type=2
            v8i_ g1;
            g1[0] = (int)(1u << 16);                          // wg_mask=0 (no cluster), data_size=2B
            g1[1] = (int)(512u << 16);                        // tensor_dim0[15:0]
            g1[2] = (int)(64u << 16);                         // tensor_dim0[31:16]=0 | tensor_dim1[15:0]
            g1[3] = (int)(512u << 16);                        // tensor_dim1[31:16]=0 | tile_dim0=512
            g1[4] = 64;                                       // tile_dim1=64 (tile_dim2=0)
            g1[5] = 512;                                      // tensor_dim0_stride[31:0]
            g1[6] = 0;
            g1[7] = 0;                                        // tensor_dim1_stride=0 (2D)
            v4i_ gz;
            gz[0] = 0; gz[1] = 0; gz[2] = 0; gz[3] = 0;
            v8i_ gz8;
            gz8[0] = 0; gz8[1] = 0; gz8[2] = 0; gz8[3] = 0;
            gz8[4] = 0; gz8[5] = 0; gz8[6] = 0; gz8[7] = 0;
            __builtin_amdgcn_tensor_load_to_lds(g0, g1, gz, gz, gz8, 0);
            __builtin_amdgcn_s_wait_tensorcnt(0);
        }
        __syncthreads();                      // tile visible to all waves

        v8f tacc[4];
        #pragma unroll
        for (int n = 0; n < 4; ++n) tacc[n] = vzero;

        const unsigned short* Arow  = &aL[wave * 16 + m][0];
        const unsigned short* Bbase = hT_in + (size_t)(b * T_ + ts) * C_ * V_;

        for (int kk = 0; kk < V_; kk += 32) {
            __builtin_prefetch(Bbase + (size_t)m * V_ + kk + 128, 0, 3);
            BF16x16 a;
            #pragma unroll
            for (int r = 0; r < 8; ++r) {
                int kp = ((r & 3) << 1) + (hlf << 3) + ((r >> 2) << 4);  // ISA 16b A 16x32 layout
                a.u[r] = *(const unsigned int*)(Arow + kk + kp);
            }
            #pragma unroll
            for (int nt = 0; nt < 4; ++nt) {
                BF16x16 bb;
                const unsigned short* Bcol = Bbase + (size_t)(nt * 16 + m) * V_ + kk + hlf * 8;
                #pragma unroll
                for (int r = 0; r < 8; ++r)
                    bb.u[r] = *(const unsigned int*)(Bcol + ((r & 3) << 1) + ((r >> 2) << 4));
                tacc[nt] = __builtin_amdgcn_wmma_f32_16x16x32_bf16(
                    false, a.v, false, bb.v, (short)0, tacc[nt], false, false);
            }
        }
        #pragma unroll
        for (int n = 0; n < 4; ++n)
            #pragma unroll
            for (int j = 0; j < 8; ++j) facc[n][j] += wf * tacc[n][j];
    }

    // C/D layout: VGPR j -> row = j + 8*(lane/16), col = nt*16 + lane%16
    #pragma unroll
    for (int n = 0; n < 4; ++n)
        #pragma unroll
        for (int j = 0; j < 8; ++j)
            aggL[wave * 16 + j + hlf * 8][n * 16 + m] = facc[n][j];
    __syncthreads();

    // Epilogue (all 128 threads; 2 per row): z = agg@xw^T + xb + SE + TE;
    // layernorm(2C) via LDS partial sums; GLU -> h (C)
    {
        int row  = tid >> 1;
        int part = tid & 1;
        int vj   = vjblk + row;
        float ar[C_];
        #pragma unroll
        for (int c = 0; c < C_; ++c) ar[c] = aggL[row][c];

        float sum = 0.f, sq = 0.f;
        for (int oo = 0; oo < C_; ++oo) {
            int o = part * C_ + oo;
            float acc = gc_xb[layer * C2_ + o] + se[vj * C2_ + o] + te[(b * T_ + t) * C2_ + o];
            #pragma unroll
            for (int c = 0; c < C_; ++c) acc += ar[c] * xwL[o][c];
            zL[row][o] = acc;
            sum += acc; sq += acc * acc;
        }
        redS[row][part] = sum;
        redQ[row][part] = sq;
        __syncthreads();
        float mean = (redS[row][0] + redS[row][1]) * (1.f / C2_);
        float var  = (redQ[row][0] + redQ[row][1]) * (1.f / C2_) - mean * mean;
        float inv  = rsqrtf(var + 1e-5f);
        for (int oo = 0; oo < C_; ++oo) {
            int o = part * C_ + oo;
            zL[row][o] = (zL[row][o] - mean) * inv * gc_ln_g[layer * C2_ + o]
                       + gc_ln_b[layer * C2_ + o];
        }
        __syncthreads();
        size_t hb  = (size_t)(b * T_ + t) * V_ * C_;
        size_t htb = (size_t)(b * T_ + t) * C_ * V_;
        for (int cc = 0; cc < 32; ++cc) {
            int c = part * 32 + cc;
            float hv = zL[row][c] * sigmoidf_(zL[row][c + C_]);
            unsigned short bv = f2bf(hv);
            h_out[hb + (size_t)vj * C_ + c]   = bv;
            hT_out[htb + (size_t)c * V_ + vj] = bv;
        }
    }
}

// ---------------------------------------------------------------------------
// GFS skip: hs[v,o] = sum_{t,c} h[b,t,v,c]*fs_w[si,o,t,c] + fs_b ; GLU(2C) -> skT bf16
__global__ __launch_bounds__(128) void k_gfs(
        const unsigned short* __restrict__ h, const unsigned short* __restrict__ fswb,
        const float* __restrict__ fs_b, const float* __restrict__ fs_glu_w,
        const float* __restrict__ fs_glu_b, int si, unsigned short* __restrict__ skT) {
    __shared__ float hsL[64][C_ + 4];
    __shared__ float gwL[C2_][C_];
    const v8f vzero = {0, 0, 0, 0, 0, 0, 0, 0};

    int b = blockIdx.y;
    int vbase = blockIdx.x * 64;
    int tid = threadIdx.x, lane = tid & 31, wave = tid >> 5;
    int m = lane & 15, hlf = lane >> 4;

    for (int i = tid; i < C2_ * C_; i += 128)
        gwL[i / C_][i % C_] = fs_glu_w[si * C2_ * C_ + i];

    v8f acc[4];
    #pragma unroll
    for (int n = 0; n < 4; ++n) acc[n] = vzero;

    const unsigned short* Abase = h + (size_t)b * T_ * V_ * C_;
    int row = vbase + wave * 16 + m;

    for (int kk = 0; kk < T_ * C_; kk += 32) {     // K = T*C = 768
        BF16x16 a;
        #pragma unroll
        for (int r = 0; r < 8; ++r) {
            int k = kk + ((r & 3) << 1) + (hlf << 3) + ((r >> 2) << 4);
            int tt = k >> 6, cc = k & 63;
            a.u[r] = *(const unsigned int*)(Abase + ((size_t)tt * V_ + row) * C_ + cc);
        }
        #pragma unroll
        for (int nt = 0; nt < 4; ++nt) {
            BF16x16 bb;
            int o = nt * 16 + m;
            #pragma unroll
            for (int r = 0; r < 8; ++r) {
                int k = kk + ((r & 3) << 1) + (hlf << 3) + ((r >> 2) << 4);
                int tt = k >> 6, cc = k & 63;
                bb.u[r] = *(const unsigned int*)(fswb + ((size_t)(si * C_ + o) * T_ + tt) * C_ + cc);
            }
            acc[nt] = __builtin_amdgcn_wmma_f32_16x16x32_bf16(
                false, a.v, false, bb.v, (short)0, acc[nt], false, false);
        }
    }
    #pragma unroll
    for (int n = 0; n < 4; ++n)
        #pragma unroll
        for (int j = 0; j < 8; ++j)
            hsL[wave * 16 + j + hlf * 8][n * 16 + m] = acc[n][j];
    __syncthreads();

    // Epilogue (2 threads per row): GLU over 2C
    {
        int r = tid >> 1, part = tid & 1;
        int v = vbase + r;
        float hs[C_];
        #pragma unroll
        for (int c = 0; c < C_; ++c) hs[c] = hsL[r][c] + fs_b[si * C_ + c];
        for (int oo = 0; oo < 32; ++oo) {
            int o2 = part * 32 + oo;
            float gl = fs_glu_b[si * C2_ + o2];
            float gr = fs_glu_b[si * C2_ + C_ + o2];
            #pragma unroll
            for (int c = 0; c < C_; ++c) {
                gl += gwL[o2][c] * hs[c];
                gr += gwL[C_ + o2][c] * hs[c];
            }
            float sv = gl * sigmoidf_(gr);
            skT[((size_t)(b * V_ + v)) * C4_ + si * C_ + o2] = f2bf(sv);
        }
    }
}

// ---------------------------------------------------------------------------
// Final: g8 = skT@glu_w^T (M=512,N=512,K=256, WMMA); GLU(8C) ; out = gg@out_w^T
__global__ __launch_bounds__(128) void k_final(
        const unsigned short* __restrict__ skT, const unsigned short* __restrict__ gluwb,
        const float* __restrict__ glu_b, const float* __restrict__ out_w,
        const float* __restrict__ out_b, float* __restrict__ out) {
    __shared__ float ggL[64][C4_ + 8];   // post-GLU activations (64x256 f32)
    const v8f vzero = {0, 0, 0, 0, 0, 0, 0, 0};

    int b = blockIdx.y;
    int vbase = blockIdx.x * 64;
    int tid = threadIdx.x, lane = tid & 31, wave = tid >> 5;
    int m = lane & 15, hlf = lane >> 4;

    const unsigned short* Abase = skT + (size_t)b * V_ * C4_;
    int row = vbase + wave * 16 + m;

    // GLU pairs column c with c+256: process (chunk pc, chunk pc+2) together
    for (int pc = 0; pc < 2; ++pc) {
        v8f accL[8], accR[8];
        #pragma unroll
        for (int n = 0; n < 8; ++n) { accL[n] = vzero; accR[n] = vzero; }
        for (int kk = 0; kk < C4_; kk += 32) {
            BF16x16 a;
            #pragma unroll
            for (int r = 0; r < 8; ++r) {
                int kp = ((r & 3) << 1) + (hlf << 3) + ((r >> 2) << 4);
                a.u[r] = *(const unsigned int*)(Abase + (size_t)row * C4_ + kk + kp);
            }
            #pragma unroll
            for (int nt = 0; nt < 8; ++nt) {
                int oL = pc * 128 + nt * 16 + m;
                int oR = oL + C4_;
                BF16x16 bL, bR;
                #pragma unroll
                for (int r = 0; r < 8; ++r) {
                    int k = kk + ((r & 3) << 1) + (hlf << 3) + ((r >> 2) << 4);
                    bL.u[r] = *(const unsigned int*)(gluwb + (size_t)oL * C4_ + k);
                    bR.u[r] = *(const unsigned int*)(gluwb + (size_t)oR * C4_ + k);
                }
                accL[nt] = __builtin_amdgcn_wmma_f32_16x16x32_bf16(
                    false, a.v, false, bL.v, (short)0, accL[nt], false, false);
                accR[nt] = __builtin_amdgcn_wmma_f32_16x16x32_bf16(
                    false, a.v, false, bR.v, (short)0, accR[nt], false, false);
            }
        }
        #pragma unroll
        for (int nt = 0; nt < 8; ++nt)
            #pragma unroll
            for (int j = 0; j < 8; ++j) {
                int col = pc * 128 + nt * 16 + m;
                float lhs = accL[nt][j] + glu_b[col];
                float rhs = accR[nt][j] + glu_b[col + C4_];
                ggL[wave * 16 + j + hlf * 8][col] = lhs * sigmoidf_(rhs);
            }
    }
    __syncthreads();

    if (tid < 64) {
        int r = tid, v = vbase + r;
        float oacc[P_];
        #pragma unroll
        for (int p = 0; p < P_; ++p) oacc[p] = out_b[p];
        for (int c = 0; c < C4_; ++c) {
            float gg = ggL[r][c];
            #pragma unroll
            for (int p = 0; p < P_; ++p) oacc[p] += gg * out_w[p * C4_ + c];
        }
        #pragma unroll
        for (int p = 0; p < P_; ++p)
            out[((size_t)(b * P_ + p)) * V_ + v] = oacc[p];
    }
}

// ---------------------------------------------------------------------------
extern "C" void kernel_launch(void* const* d_in, const int* in_sizes, int n_in,
                              void* d_out, int out_size, void* d_ws, size_t ws_size,
                              hipStream_t stream) {
    (void)in_sizes; (void)n_in; (void)out_size; (void)ws_size;
    const float* x         = (const float*)d_in[0];
    const float* sape      = (const float*)d_in[1];
    const float* tape      = (const float*)d_in[2];
    const float* srpe      = (const float*)d_in[3];
    const float* trpe      = (const float*)d_in[4];
    // d_in[5], d_in[6] are explicit zero pads; handled analytically (tau skip)
    const float* range_mask= (const float*)d_in[7];
    const float* W_in      = (const float*)d_in[8];
    const float* b_in      = (const float*)d_in[9];
    const float* fs_w      = (const float*)d_in[10];
    const float* fs_b      = (const float*)d_in[11];
    const float* fs_glu_w  = (const float*)d_in[12];
    const float* fs_glu_b  = (const float*)d_in[13];
    const float* ri_mu     = (const float*)d_in[14];
    const float* ri_sigma  = (const float*)d_in[15];
    const float* gc_xw     = (const float*)d_in[16];
    const float* gc_xb     = (const float*)d_in[17];
    const float* gc_sew    = (const float*)d_in[18];
    const float* gc_tew    = (const float*)d_in[19];
    const float* gc_ln_g   = (const float*)d_in[20];
    const float* gc_ln_b   = (const float*)d_in[21];
    const float* glu_w     = (const float*)d_in[22];
    const float* glu_b     = (const float*)d_in[23];
    const float* out_w     = (const float*)d_in[24];
    const float* out_b     = (const float*)d_in[25];
    float* out = (float*)d_out;

    char* ws = (char*)d_ws;
    size_t off = 0;
    auto walloc = [&](size_t bytes) -> void* {
        void* p = ws + off;
        off += (bytes + 255) & ~(size_t)255;
        return p;
    };
    unsigned short* hA    = (unsigned short*)walloc((size_t)BTVC * 2);
    unsigned short* hB    = (unsigned short*)walloc((size_t)BTVC * 2);
    unsigned short* hTA   = (unsigned short*)walloc((size_t)BTVC * 2);
    unsigned short* hTB   = (unsigned short*)walloc((size_t)BTVC * 2);
    unsigned short* emask = (unsigned short*)walloc((size_t)TS_ * V_ * V_ * 2);
    float*          wfac  = (float*)walloc((size_t)B_ * T_ * TS_ * 4);
    float*          se    = (float*)walloc((size_t)V_ * C2_ * 4);
    float*          te    = (float*)walloc((size_t)B_ * T_ * C2_ * 4);
    unsigned short* fswb  = (unsigned short*)walloc((size_t)(L_ + 1) * C_ * T_ * C_ * 2);
    unsigned short* gluwb = (unsigned short*)walloc((size_t)(8 * C_) * C4_ * 2);
    unsigned short* skT   = (unsigned short*)walloc((size_t)B_ * V_ * C4_ * 2);

    const int nFsw  = (L_ + 1) * C_ * T_ * C_;   // 196608
    const int nGluw = 8 * C_ * C4_;              // 131072

    k_prep_h<<<(BTVC + 255) / 256, 256, 0, stream>>>(x, W_in, b_in, hA, hTA);
    k_f2bf_arr<<<(nFsw + 255) / 256, 256, 0, stream>>>(fs_w, fswb, nFsw);
    k_f2bf_arr<<<(nGluw + 255) / 256, 256, 0, stream>>>(glu_w, gluwb, nGluw);

    // skip[0] from h0
    k_gfs<<<dim3(V_ / 64, B_), 128, 0, stream>>>(hA, fswb, fs_b, fs_glu_w, fs_glu_b, 0, skT);

    unsigned short *ch = hA, *chT = hTA, *nh = hB, *nhT = hTB;
    for (int i = 0; i < L_; ++i) {
        k_emask<<<(V_ * V_ + 255) / 256, 256, 0, stream>>>(sape, srpe, range_mask,
                                                           ri_mu, ri_sigma, i, emask);
        k_wfac<<<1, 256, 0, stream>>>(tape, trpe, ri_mu, ri_sigma, i, wfac);
        k_se<<<(V_ * C2_ + 255) / 256, 256, 0, stream>>>(sape, gc_sew, i, se);
        k_te<<<(B_ * T_ * C2_ + 255) / 256, 256, 0, stream>>>(tape, gc_tew, i, te);
        k_layer<<<dim3(V_ / 64, T_, B_), 128, 0, stream>>>(emask, chT, wfac, se, te,
                                                           gc_xw, gc_xb, gc_ln_g, gc_ln_b,
                                                           i, nh, nhT);
        k_gfs<<<dim3(V_ / 64, B_), 128, 0, stream>>>(nh, fswb, fs_b, fs_glu_w, fs_glu_b,
                                                     i + 1, skT);
        unsigned short* tmp;
        tmp = ch;  ch  = nh;  nh  = tmp;
        tmp = chT; chT = nhT; nhT = tmp;
    }

    k_final<<<dim3(V_ / 64, B_), 128, 0, stream>>>(skT, gluwb, glu_b, out_w, out_b, out);
}
